// MoEMLP_63840393888110
// MI455X (gfx1250) — matrix operand order repaired and checked by reading, not compiled, not run on hardware
//
#include <hip/hip_runtime.h>
#include <hip/hip_bf16.h>
#include <math.h>

typedef __bf16 bf16;
typedef __attribute__((ext_vector_type(16))) __bf16 bf16x16;
typedef __attribute__((ext_vector_type(8)))  float   f32x8;

namespace {
constexpr int T_TOK = 32768;   // B*S tokens
constexpr int D_DIM = 1024;
constexpr int E_NUM = 16;
constexpr int I_DIM = 2816;
constexpr int I2    = 5632;    // 2*I
constexpr int TK    = 65536;   // T*k
constexpr int CAP   = 5120;    // ceil(T*k/E*1.25)
}

union Frag  { bf16x16 v; uint4 q[2]; };
union Pack8 { uint4 q; bf16 h[8]; };

// ---------------------------------------------------------------------------
// CDNA5 async global->LDS copy (ASYNCcnt-tracked), GVS addressing:
//   mem = SADDR(64b SGPR pair) + VADDR(32b VGPR) ; LDS dst = VDST VGPR offset
// ---------------------------------------------------------------------------
__device__ __forceinline__ void async_g2l_b128(unsigned lds_off, unsigned voff,
                                               const void* sbase) {
    asm volatile("global_load_async_to_lds_b128 %0, %1, %2"
                 :
                 : "v"(lds_off), "v"(voff), "s"(sbase)
                 : "memory");
}
__device__ __forceinline__ void wait_async0() {
    asm volatile("s_wait_asynccnt 0x0" ::: "memory");
}
__device__ __forceinline__ unsigned lds_off_u32(const void* p) {
    // generic LDS pointer: low 32 bits are the wave-relative LDS byte address
    return (unsigned)(uintptr_t)p;
}

// ---------------------------------------------------------------------------
// 0) zero small stats scratch (prob_sum[16], z2sum[1], counts_i[16])
// ---------------------------------------------------------------------------
__global__ void moe_zero_stats(unsigned* stats) {
    if (threadIdx.x < 33) stats[threadIdx.x] = 0u;
}

// ---------------------------------------------------------------------------
// 1) router: logits, softmax stats, top-2 selection, combine weights.
//    One wave (32 lanes) per token; 8 tokens per 256-thread block.
// ---------------------------------------------------------------------------
__global__ void __launch_bounds__(256)
moe_router(const float* __restrict__ x, const float* __restrict__ router,
           int* __restrict__ sel, float* __restrict__ comb,
           float* __restrict__ prob_sum, float* __restrict__ z2sum)
{
    __shared__ float s_prob[E_NUM];
    __shared__ float s_z2;
    const int tid = threadIdx.x;
    if (tid < E_NUM) s_prob[tid] = 0.f;
    if (tid == 0)    s_z2 = 0.f;
    __syncthreads();

    const int wid  = tid >> 5;
    const int lane = tid & 31;
    const int t = blockIdx.x * 8 + wid;

    float acc[E_NUM];
#pragma unroll
    for (int e = 0; e < E_NUM; ++e) acc[e] = 0.f;

    const float* xr = x + (size_t)t * D_DIM;
    for (int d = lane; d < D_DIM; d += 32) {
        const float xv = xr[d];
        const float4* r4 = reinterpret_cast<const float4*>(router + (size_t)d * E_NUM);
#pragma unroll
        for (int j = 0; j < 4; ++j) {
            const float4 rv = r4[j];
            acc[4 * j + 0] += xv * rv.x;
            acc[4 * j + 1] += xv * rv.y;
            acc[4 * j + 2] += xv * rv.z;
            acc[4 * j + 3] += xv * rv.w;
        }
    }
#pragma unroll
    for (int off = 16; off >= 1; off >>= 1) {
#pragma unroll
        for (int e = 0; e < E_NUM; ++e) acc[e] += __shfl_down(acc[e], off, 32);
    }

    if (lane == 0) {
        float m = acc[0]; int i1 = 0;
#pragma unroll
        for (int e = 1; e < E_NUM; ++e) if (acc[e] > m) { m = acc[e]; i1 = e; }
        float pr[E_NUM]; float sum = 0.f;
#pragma unroll
        for (int e = 0; e < E_NUM; ++e) { pr[e] = __expf(acc[e] - m); sum += pr[e]; }
        const float inv = 1.f / sum;
#pragma unroll
        for (int e = 0; e < E_NUM; ++e) atomicAdd(&s_prob[e], pr[e] * inv);
        const float z = m + __logf(sum);           // logsumexp
        atomicAdd(&s_z2, z * z);

        float m2 = -1e30f; int i2 = 0;
#pragma unroll
        for (int e = 0; e < E_NUM; ++e) if (e != i1 && acc[e] > m2) { m2 = acc[e]; i2 = e; }
        const float c0 = 1.f / (1.f + __expf(m2 - m)); // softmax over top-2
        sel[2 * t]     = i1;
        sel[2 * t + 1] = i2;
        comb[2 * t]     = c0;
        comb[2 * t + 1] = 1.f - c0;
    }
    __syncthreads();
    if (tid < E_NUM) atomicAdd(&prob_sum[tid], s_prob[tid]);
    if (tid == 0)    atomicAdd(z2sum, s_z2);
}

// ---------------------------------------------------------------------------
// 2) exact capacity-position assignment (replicates the reference cumsum):
//    single block, per-expert exclusive scan over 1024 contiguous chunks.
// ---------------------------------------------------------------------------
__global__ void __launch_bounds__(1024)
moe_assign_pos(const int* __restrict__ sel, int* __restrict__ pos,
               int* __restrict__ counts_i)
{
    __shared__ int buf[1024];
    const int tid = threadIdx.x;
    const int CH = TK / 1024;            // 64 entries per thread
    const int base = tid * CH;

    int local[E_NUM];
#pragma unroll
    for (int e = 0; e < E_NUM; ++e) local[e] = 0;
    for (int i = 0; i < CH; ++i) local[sel[base + i]]++;

    int start[E_NUM];
    for (int e = 0; e < E_NUM; ++e) {
        buf[tid] = local[e];
        __syncthreads();
        for (int off = 1; off < 1024; off <<= 1) {
            const int v = (tid >= off) ? buf[tid - off] : 0;
            __syncthreads();
            buf[tid] += v;
            __syncthreads();
        }
        start[e] = (tid > 0) ? buf[tid - 1] : 0;
        if (tid == 1023) counts_i[e] = buf[1023];
        __syncthreads();
    }
    for (int i = 0; i < CH; ++i) {
        const int e = sel[base + i];
        const int p = start[e]++;
        pos[base + i] = (p < CAP) ? p : -1;  // -1 => dropped
    }
}

// ---------------------------------------------------------------------------
// 3) scatter tokens into expert_in (bf16), one block per kept entry
// ---------------------------------------------------------------------------
__global__ void __launch_bounds__(128)
moe_scatter(const float* __restrict__ x, const int* __restrict__ sel,
            const int* __restrict__ pos, bf16* __restrict__ ein)
{
    const int entry = blockIdx.x;
    const int p = pos[entry];
    if (p < 0) return;
    const int e = sel[entry];
    const int t = entry >> 1;
    const int tid = threadIdx.x;           // 128 threads * 8 elems = 1024
    const float4* src = reinterpret_cast<const float4*>(x + (size_t)t * D_DIM) + tid * 2;
    const float4 a = src[0];
    const float4 b = src[1];
    Pack8 o;
    o.h[0] = (bf16)a.x; o.h[1] = (bf16)a.y; o.h[2] = (bf16)a.z; o.h[3] = (bf16)a.w;
    o.h[4] = (bf16)b.x; o.h[5] = (bf16)b.y; o.h[6] = (bf16)b.z; o.h[7] = (bf16)b.w;
    uint4* dst = reinterpret_cast<uint4*>(ein + ((size_t)e * CAP + p) * D_DIM) + tid;
    *dst = o.q;
}

// ---------------------------------------------------------------------------
// 4) fused up+gate GEMM + SiLU*up epilogue.
//    Tile: 128(M) x (64 gate | 64 up)(N), K-step 32 (one WMMA K).
//    A tile staged via async global->LDS DMA; B tiles cvt fp32->bf16 on VALU.
// ---------------------------------------------------------------------------
__global__ void __launch_bounds__(256)
moe_gemm_up_act(const bf16* __restrict__ ein, const float* __restrict__ wug,
                bf16* __restrict__ act)
{
    __shared__ __align__(16) bf16 As[128 * 32];   // [M][K] row-major
    __shared__ __align__(16) bf16 Bg[64 * 32];    // [N][K] (transposed)
    __shared__ __align__(16) bf16 Bu[64 * 32];

    const int e  = blockIdx.z;
    const int m0 = blockIdx.y * 128;
    const int n0 = blockIdx.x * 64;
    const int tid  = threadIdx.x;
    const int wid  = tid >> 5;
    const int lane = tid & 31;
    const int waveM = wid >> 2;
    const int waveN = wid & 3;
    const int lrow  = lane & 15;
    const int lhi   = (lane >> 4) & 1;

    f32x8 accG[4] = {};
    f32x8 accU[4] = {};

    const bf16*  Abase = ein + ((size_t)e * CAP + m0) * D_DIM;
    const float* Bbase = wug + (size_t)e * D_DIM * I2;
    const unsigned as_lds = lds_off_u32(&As[0]);

    for (int k0 = 0; k0 < D_DIM; k0 += 32) {
        // stage A tile (bf16 128x32) with async DMA: global -> LDS direct
#pragma unroll
        for (int j = 0; j < 2; ++j) {
            const int idx = tid + j * 256;
            const int r = idx >> 2;
            const int c = (idx & 3) * 8;
            async_g2l_b128(as_lds + (unsigned)(r * 32 + c) * 2u,
                           (unsigned)(((size_t)r * D_DIM + (k0 + c)) * 2u),
                           (const void*)Abase);
        }
        // stage B tiles: fp32 -> bf16 on the fly, stored [N][K]
#pragma unroll
        for (int j = 0; j < 2; ++j) {
            const int idx = tid + j * 256;
            const int lin = idx * 4;
            const int kk = lin >> 6;
            const int n  = lin & 63;
            const float* rowp = Bbase + (size_t)(k0 + kk) * I2 + n0 + n;
            const float4 gv = *reinterpret_cast<const float4*>(rowp);
            const float4 uv = *reinterpret_cast<const float4*>(rowp + I_DIM);
            Bg[(n + 0) * 32 + kk] = (bf16)gv.x;
            Bg[(n + 1) * 32 + kk] = (bf16)gv.y;
            Bg[(n + 2) * 32 + kk] = (bf16)gv.z;
            Bg[(n + 3) * 32 + kk] = (bf16)gv.w;
            Bu[(n + 0) * 32 + kk] = (bf16)uv.x;
            Bu[(n + 1) * 32 + kk] = (bf16)uv.y;
            Bu[(n + 2) * 32 + kk] = (bf16)uv.z;
            Bu[(n + 3) * 32 + kk] = (bf16)uv.w;
        }
        wait_async0();
        __syncthreads();

        Frag bg, bu;
        const int nrow = (waveN * 16 + lrow) * 32;
        const int kb = lhi * 16;               // lanes16-31 hold K 16..31
        bg.q[0] = *reinterpret_cast<const uint4*>(&Bg[nrow + kb]);
        bg.q[1] = *reinterpret_cast<const uint4*>(&Bg[nrow + kb + 8]);
        bu.q[0] = *reinterpret_cast<const uint4*>(&Bu[nrow + kb]);
        bu.q[1] = *reinterpret_cast<const uint4*>(&Bu[nrow + kb + 8]);
        const int ka = lhi * 16;               // lanes16-31: K 16..31
#pragma unroll
        for (int mf = 0; mf < 4; ++mf) {
            Frag a;
            const int ar = (waveM * 64 + mf * 16 + lrow) * 32;
            a.q[0] = *reinterpret_cast<const uint4*>(&As[ar + ka]);
            a.q[1] = *reinterpret_cast<const uint4*>(&As[ar + ka + 8]);
            accG[mf] = __builtin_amdgcn_wmma_f32_16x16x32_bf16(
                false, a.v, false, bg.v, (short)0, accG[mf], false, false);
            accU[mf] = __builtin_amdgcn_wmma_f32_16x16x32_bf16(
                false, a.v, false, bu.v, (short)0, accU[mf], false, false);
        }
        __syncthreads();
    }

    const int col = n0 + waveN * 16 + lrow;
#pragma unroll
    for (int mf = 0; mf < 4; ++mf) {
        const int rbase = m0 + waveM * 64 + mf * 16 + lhi * 8;
#pragma unroll
        for (int i = 0; i < 8; ++i) {
            const float g = accG[mf][i];
            const float u = accU[mf][i];
            // silu via hardware v_rcp_f32 (avoid IEEE divide expansion)
            const float s = g * __builtin_amdgcn_rcpf(1.f + __expf(-g));
            act[((size_t)e * CAP + rbase + i) * I_DIM + col] = (bf16)(s * u);
        }
    }
}

// ---------------------------------------------------------------------------
// 5) down GEMM: act(C x I) @ w_down(I x D) -> expert_out bf16.
//    Tile 128x128, 8 waves = 2(M) x 4(N), each wave 4 M-frags x 2 N-frags.
// ---------------------------------------------------------------------------
__global__ void __launch_bounds__(256)
moe_gemm_down(const bf16* __restrict__ act, const float* __restrict__ wdn,
              bf16* __restrict__ eout)
{
    __shared__ __align__(16) bf16 As[128 * 32];
    __shared__ __align__(16) bf16 Bs[128 * 32];   // [N][K]

    const int e  = blockIdx.z;
    const int m0 = blockIdx.y * 128;
    const int n0 = blockIdx.x * 128;
    const int tid  = threadIdx.x;
    const int wid  = tid >> 5;
    const int lane = tid & 31;
    const int waveM = wid >> 2;
    const int waveN = wid & 3;
    const int lrow  = lane & 15;
    const int lhi   = (lane >> 4) & 1;

    f32x8 acc[4][2] = {};

    const bf16*  Abase = act + ((size_t)e * CAP + m0) * I_DIM;
    const float* Bbase = wdn + (size_t)e * I_DIM * D_DIM;
    const unsigned as_lds = lds_off_u32(&As[0]);

    for (int k0 = 0; k0 < I_DIM; k0 += 32) {
        // stage A tile via async global->LDS DMA
#pragma unroll
        for (int j = 0; j < 2; ++j) {
            const int idx = tid + j * 256;
            const int r = idx >> 2;
            const int c = (idx & 3) * 8;
            async_g2l_b128(as_lds + (unsigned)(r * 32 + c) * 2u,
                           (unsigned)(((size_t)r * I_DIM + (k0 + c)) * 2u),
                           (const void*)Abase);
        }
#pragma unroll
        for (int j = 0; j < 4; ++j) {
            const int idx = tid + j * 256;
            const int lin = idx * 4;
            const int kk = lin >> 7;
            const int n  = lin & 127;
            const float4 v = *reinterpret_cast<const float4*>(
                Bbase + (size_t)(k0 + kk) * D_DIM + n0 + n);
            Bs[(n + 0) * 32 + kk] = (bf16)v.x;
            Bs[(n + 1) * 32 + kk] = (bf16)v.y;
            Bs[(n + 2) * 32 + kk] = (bf16)v.z;
            Bs[(n + 3) * 32 + kk] = (bf16)v.w;
        }
        wait_async0();
        __syncthreads();

        const int kb = lhi * 16;
        const int ka = lhi * 16;
        Frag b[2];
#pragma unroll
        for (int nf = 0; nf < 2; ++nf) {
            const int nrow = (waveN * 32 + nf * 16 + lrow) * 32;
            b[nf].q[0] = *reinterpret_cast<const uint4*>(&Bs[nrow + kb]);
            b[nf].q[1] = *reinterpret_cast<const uint4*>(&Bs[nrow + kb + 8]);
        }
#pragma unroll
        for (int mf = 0; mf < 4; ++mf) {
            Frag a;
            const int ar = (waveM * 64 + mf * 16 + lrow) * 32;
            a.q[0] = *reinterpret_cast<const uint4*>(&As[ar + ka]);
            a.q[1] = *reinterpret_cast<const uint4*>(&As[ar + ka + 8]);
#pragma unroll
            for (int nf = 0; nf < 2; ++nf)
                acc[mf][nf] = __builtin_amdgcn_wmma_f32_16x16x32_bf16(
                    false, a.v, false, b[nf].v, (short)0, acc[mf][nf], false, false);
        }
        __syncthreads();
    }

#pragma unroll
    for (int mf = 0; mf < 4; ++mf) {
        const int rbase = m0 + waveM * 64 + mf * 16 + lhi * 8;
#pragma unroll
        for (int nf = 0; nf < 2; ++nf) {
            const int col = n0 + waveN * 32 + nf * 16 + lrow;
#pragma unroll
            for (int i = 0; i < 8; ++i)
                eout[((size_t)e * CAP + rbase + i) * D_DIM + col] = (bf16)acc[mf][nf][i];
        }
    }
}

// ---------------------------------------------------------------------------
// 6) gather + combine-weighted sum into routed output
// ---------------------------------------------------------------------------
__global__ void __launch_bounds__(128)
moe_gather(const bf16* __restrict__ eout, const int* __restrict__ sel,
           const int* __restrict__ pos, const float* __restrict__ comb,
           float* __restrict__ outp)
{
    const int t = blockIdx.x;
    const int tid = threadIdx.x;
    const int d0 = tid * 8;
    float a[8];
#pragma unroll
    for (int i = 0; i < 8; ++i) a[i] = 0.f;
#pragma unroll
    for (int j = 0; j < 2; ++j) {
        const int entry = 2 * t + j;
        const int p = pos[entry];
        if (p >= 0) {
            const int e = sel[entry];
            const float w = comb[entry];
            Pack8 pk;
            pk.q = *reinterpret_cast<const uint4*>(
                eout + ((size_t)e * CAP + p) * D_DIM + d0);
#pragma unroll
            for (int i = 0; i < 8; ++i) a[i] += w * (float)pk.h[i];
        }
    }
    float4* dst = reinterpret_cast<float4*>(outp + (size_t)t * D_DIM + d0);
    dst[0] = make_float4(a[0], a[1], a[2], a[3]);
    dst[1] = make_float4(a[4], a[5], a[6], a[7]);
}

// ---------------------------------------------------------------------------
// 7) finalize aux stats into d_out tail: counts[16], entropy, lb_loss, z_loss
// ---------------------------------------------------------------------------
__global__ void moe_finalize(const int* __restrict__ counts_i,
                             const float* __restrict__ prob_sum,
                             const float* __restrict__ z2sum,
                             float* __restrict__ tail)
{
    if (threadIdx.x != 0) return;
    float cnt[E_NUM]; float total = 0.f;
    for (int e = 0; e < E_NUM; ++e) { cnt[e] = (float)counts_i[e]; total += cnt[e]; }
    total = fmaxf(total, 1.f);
    float ent = 0.f, lb = 0.f;
    for (int e = 0; e < E_NUM; ++e) {
        const float fr = cnt[e] / total;
        ent -= fr * logf(fr + 1e-6f);
        lb += (fr * 2.f) * (prob_sum[e] / (float)T_TOK);   // token_fraction * p_mean
    }
    lb *= (float)E_NUM;
    for (int e = 0; e < E_NUM; ++e) tail[e] = cnt[e];
    tail[16] = ent;
    tail[17] = lb;
    tail[18] = z2sum[0] / (float)T_TOK;
}

// ---------------------------------------------------------------------------
// launch
// ---------------------------------------------------------------------------
extern "C" void kernel_launch(void* const* d_in, const int* in_sizes, int n_in,
                              void* d_out, int out_size, void* d_ws, size_t ws_size,
                              hipStream_t stream) {
    (void)in_sizes; (void)n_in; (void)out_size; (void)ws_size;
    const float* x      = (const float*)d_in[0];   // (8,4096,1024)
    const float* router = (const float*)d_in[1];   // (1024,16)
    const float* wug    = (const float*)d_in[2];   // (16,1024,5632)
    const float* wdn    = (const float*)d_in[3];   // (16,2816,1024)
    // d_in[4] = num_experts_per_token (== 2, hardcoded)

    char* ws = (char*)d_ws;
    int*   sel      = (int*)  (ws + 0);                   // TK ints
    float* comb     = (float*)(ws + 262144);              // TK floats
    int*   pos      = (int*)  (ws + 524288);              // TK ints
    float* prob_sum = (float*)(ws + 786432);              // 16 f32
    float* z2sum    = prob_sum + 16;                      // 1  f32
    int*   counts_i = (int*)(prob_sum + 17);              // 16 i32
    bf16*  ein      = (bf16*)(ws + 1048576ULL);           // 16*5120*1024 bf16
    bf16*  act      = (bf16*)(ws + 168820736ULL);         // 16*5120*2816 bf16
    bf16*  eout     = (bf16*)(ws + 630194176ULL);         // 16*5120*1024 bf16
    float* outp = (float*)d_out;                          // routed (T,D)
    float* tail = outp + (size_t)T_TOK * D_DIM;           // counts/entropy/lb/z

    moe_zero_stats<<<1, 64, 0, stream>>>((unsigned*)prob_sum);
    moe_router<<<T_TOK / 8, 256, 0, stream>>>(x, router, sel, comb, prob_sum, z2sum);
    moe_assign_pos<<<1, 1024, 0, stream>>>(sel, pos, counts_i);
    moe_scatter<<<TK, 128, 0, stream>>>(x, sel, pos, ein);
    moe_gemm_up_act<<<dim3(I_DIM / 64, CAP / 128, E_NUM), 256, 0, stream>>>(ein, wug, act);
    moe_gemm_down<<<dim3(D_DIM / 128, CAP / 128, E_NUM), 256, 0, stream>>>(act, wdn, eout);
    moe_gather<<<T_TOK, 128, 0, stream>>>(eout, sel, pos, comb, outp);
    moe_finalize<<<1, 32, 0, stream>>>(counts_i, prob_sum, z2sum, tail);
}
